// SFNAttention_11793980195019
// MI455X (gfx1250) — matrix operand order repaired and checked by reading, not compiled, not run on hardware
//
#include <hip/hip_runtime.h>
#include <hip/hip_bf16.h>

// ---------------------------------------------------------------------------
// SFN attention on gfx1250: f16 WMMA pipeline, f32 accumulation.
// - Fixed-shift softmax (scores provably bounded by 8 after sfn + qk_norm).
// - qk_norm fused into the Q/K projection epilogue (wave tile == head tile).
// - V projection writes V^T so the PV operand is contiguous everywhere.
// - Flash attention: 8 waves/block share double-buffered K/V^T chunks staged
//   with global_load_async_to_lds_b128 (ASYNCcnt pipelining).
// ---------------------------------------------------------------------------

typedef __attribute__((ext_vector_type(16))) _Float16 v16h;
typedef __attribute__((ext_vector_type(8)))  _Float16 v8h;
typedef __attribute__((ext_vector_type(8)))  float    v8f;

#define D_MODEL 1024
#define NHEAD   16
#define HEAD_DIM 64
#define SEQ     2048
#define BATCH   2
#define ROWS    (BATCH * SEQ)        // 4096
#define SF_STEP 0.5f
#define SF_LEVELS 8.0f
#define SMAX    8.0f                 // score upper bound: 8*8*Dh^-0.5

// ------------------------------- f32 -> f16 --------------------------------
__global__ void cvt_f32_to_f16(const float* __restrict__ in,
                               _Float16* __restrict__ out, int n) {
    int i = blockIdx.x * blockDim.x + threadIdx.x;
    if (i < n) out[i] = (_Float16)in[i];
}

__device__ __forceinline__ float sfn_quant(float y) {
    return fminf(fmaxf(rintf(y * (1.0f / SF_STEP)), -SF_LEVELS), SF_LEVELS)
           * SF_STEP;
}

// --------------------- WMMA GEMM: Y = X @ W^T + bias -----------------------
// One 16x64 output tile per wave (A fragment reused across 4 WMMAs/k-step).
// The 64-wide supertile is exactly one attention head.
// MODE 1: sfn quantize, store f16 TRANSPOSED            (V projection -> V^T)
// MODE 2: store f32 row-major                           (output projection)
// MODE 3: sfn quantize + fused head RMS-norm, f16       (Q, K projections)
template <int MODE>
__global__ void wmma_gemm_bias(const _Float16* __restrict__ X,
                               const _Float16* __restrict__ W,
                               const float* __restrict__ bias,
                               const float* __restrict__ gain,   // MODE 3
                               _Float16* __restrict__ Yh,
                               float* __restrict__ Yf,
                               int Mrows) {
    const int lane    = threadIdx.x & 31;
    const int wave    = threadIdx.x >> 5;
    const int tileId  = blockIdx.x * (blockDim.x >> 5) + wave;
    const int tilesN  = D_MODEL / 64;                 // 16 supertiles of 64
    const int tileM   = tileId / tilesN;
    const int tileN   = tileId % tilesN;
    if (tileM * 16 >= Mrows) return;                  // wave-uniform

    const int row0 = tileM * 16, col0 = tileN * 64;
    const int m    = lane & 15;                       // A row / B col / C col
    const int hs   = lane >> 4;                       // half-wave select

    const _Float16* xrow = X + (size_t)(row0 + m) * D_MODEL;
    const _Float16* wrow[4];
    #pragma unroll
    for (int c = 0; c < 4; ++c)
        wrow[c] = W + (size_t)(col0 + c * 16 + m) * D_MODEL;

    v8f acc[4] = {v8f{}, v8f{}, v8f{}, v8f{}};
    const int ca = hs * 16;   // A-frag K base offset
    const int cb = hs * 16;   // B-frag K base offset
    for (int k0 = 0; k0 < D_MODEL; k0 += 32) {
        __builtin_prefetch(xrow + k0 + 256, 0, 0);    // global_prefetch_b8
        v16h av = *(const v16h*)(xrow + k0 + ca);
        #pragma unroll
        for (int c = 0; c < 4; ++c) {
            v16h bv = *(const v16h*)(wrow[c] + k0 + cb);
            acc[c] = __builtin_amdgcn_wmma_f32_16x16x32_f16(
                false, av, false, bv, (short)0, acc[c], false, false);
        }
    }

    if (MODE == 3) {
        // sfn quantize, then RMS-normalize each 64-dim head row in-register.
        float qv[4][8];
        float ss[8];
        #pragma unroll
        for (int r = 0; r < 8; ++r) ss[r] = 0.0f;
        #pragma unroll
        for (int c = 0; c < 4; ++c) {
            const float bn = bias[col0 + c * 16 + m];
            #pragma unroll
            for (int r = 0; r < 8; ++r) {
                const float q = sfn_quant(acc[c][r] + bn);
                qv[c][r] = q;
                ss[r] += q * q;
            }
        }
        float inv[8];
        #pragma unroll
        for (int r = 0; r < 8; ++r) {
            float s = ss[r];                 // reduce across 16-lane half
            #pragma unroll
            for (int off = 8; off >= 1; off >>= 1) s += __shfl_xor(s, off, 32);
            inv[r] = rsqrtf(s * (1.0f / HEAD_DIM) + 1e-6f);
        }
        #pragma unroll
        for (int c = 0; c < 4; ++c) {
            const float gn = gain[c * 16 + m];        // dh within head
            #pragma unroll
            for (int r = 0; r < 8; ++r) {
                const int Mr = r + 8 * hs;
                Yh[(size_t)(row0 + Mr) * D_MODEL + col0 + c * 16 + m] =
                    (_Float16)(qv[c][r] * inv[r] * gn);
            }
        }
    } else {
        #pragma unroll
        for (int c = 0; c < 4; ++c) {
            const int col = col0 + c * 16 + m;
            const float bn = bias[col];
            if (MODE == 1) {
                // Transposed f16 store: lane's 8 rows consecutive -> one v8h
                v8h pack;
                #pragma unroll
                for (int r = 0; r < 8; ++r)
                    pack[r] = (_Float16)sfn_quant(acc[c][r] + bn);
                *(v8h*)(Yh + (size_t)col * Mrows + row0 + 8 * hs) = pack;
            } else {
                #pragma unroll
                for (int r = 0; r < 8; ++r) {
                    const int Mr = r + 8 * hs;
                    Yf[(size_t)(row0 + Mr) * D_MODEL + col] = acc[c][r] + bn;
                }
            }
        }
    }
}

// -------------------------- flash attention --------------------------------
// grid = (SEQ/128, BATCH*NHEAD), block = 256 (8 waves).
// Each wave owns one 16-row q-tile; the 8 waves share each 32-key K / V^T
// chunk, staged into double-buffered LDS via global_load_async_to_lds_b128.
__global__ void flash_attn_kernel(const _Float16* __restrict__ Q,
                                  const _Float16* __restrict__ K,
                                  const _Float16* __restrict__ Vt,
                                  _Float16* __restrict__ O) {
    __shared__ __attribute__((aligned(32))) _Float16 ldsK [2][32 * HEAD_DIM];
    __shared__ __attribute__((aligned(32))) _Float16 ldsVt[2][HEAD_DIM * 32];
    __shared__ __attribute__((aligned(32))) _Float16 ldsP [8][16 * 32];

    const int lane = threadIdx.x & 31;
    const int wave = threadIdx.x >> 5;
    const int bh   = blockIdx.y;
    const int b    = bh >> 4, h = bh & 15;
    const size_t rowBase = (size_t)b * SEQ;
    const int q0 = (blockIdx.x * 8 + wave) * 16;
    const int m  = lane & 15;
    const int hs = lane >> 4;
    const float scale = 0.125f;                       // 64^-0.5

    // Cooperative stage of one 32-key chunk: 256 threads x 16 B each for K
    // and for V^T (each wave issues exactly 2 async instructions).
    const int sidx = wave * 32 + lane;                // 0..255
    const int krow = sidx >> 3, kcol = (sidx & 7) * 8;
    const int vrow = sidx >> 2, vcol = (sidx & 3) * 8;
    auto stage = [&](int kcN, int buf) {
        const unsigned ldsKa =
            (unsigned)(size_t)(&ldsK[buf][krow * HEAD_DIM + kcol]);
        const unsigned long long gaK = (unsigned long long)(size_t)(
            K + (rowBase + kcN + krow) * D_MODEL + h * HEAD_DIM + kcol);
        asm volatile("global_load_async_to_lds_b128 %0, %1, off"
                     :: "v"(ldsKa), "v"(gaK) : "memory");
        const unsigned ldsVa =
            (unsigned)(size_t)(&ldsVt[buf][vrow * 32 + vcol]);
        const unsigned long long gaV = (unsigned long long)(size_t)(
            Vt + (size_t)(h * HEAD_DIM + vrow) * ROWS + rowBase + kcN + vcol);
        asm volatile("global_load_async_to_lds_b128 %0, %1, off"
                     :: "v"(ldsVa), "v"(gaV) : "memory");
    };

    // Preload this wave's Q A-fragments for Dh chunks k0=0 and k0=32
    const _Float16* qrow = Q + (rowBase + q0 + m) * D_MODEL + h * HEAD_DIM;
    v16h aQ[2];
    #pragma unroll
    for (int t = 0; t < 2; ++t) {
        const int base = t * 32 + hs * 16;
        aQ[t] = *(const v16h*)(qrow + base);
    }

    float lsum[8];
    v8f accO[4] = {v8f{}, v8f{}, v8f{}, v8f{}};
    #pragma unroll
    for (int r = 0; r < 8; ++r) lsum[r] = 0.0f;

    stage(0, 0);                                      // prologue: chunk 0

    for (int kc = 0; kc < SEQ; kc += 32) {
        const int cur = (kc >> 5) & 1;
        if (kc + 32 < SEQ) {
            stage(kc + 32, cur ^ 1);                  // prefetch next chunk
            asm volatile("s_wait_asynccnt 2" ::: "memory"); // cur landed
        } else {
            asm volatile("s_wait_asynccnt 0" ::: "memory");
        }
        __syncthreads();                              // all waves' stages seen

        // Batch all K-fragment LDS loads, then the 4 score WMMAs
        v16h kb[2][2];
        #pragma unroll
        for (int t = 0; t < 2; ++t) {
            const _Float16* kr = &ldsK[cur][(t * 16 + m) * HEAD_DIM + hs * 16];
            kb[t][0] = *(const v16h*)(kr);            // Dh 0..31
            kb[t][1] = *(const v16h*)(kr + 32);       // Dh 32..63
        }
        v8f S[2];
        #pragma unroll
        for (int t = 0; t < 2; ++t) {
            v8f s = {};
            s = __builtin_amdgcn_wmma_f32_16x16x32_f16(
                false, aQ[0], false, kb[t][0], (short)0, s, false, false);
            s = __builtin_amdgcn_wmma_f32_16x16x32_f16(
                false, aQ[1], false, kb[t][1], (short)0, s, false, false);
            S[t] = s;
        }

        // Fixed-shift exp; accumulate per-lane row sums; scatter P to LDS
        #pragma unroll
        for (int r = 0; r < 8; ++r) {
            const float p0 = __expf(S[0][r] * scale - SMAX);
            const float p1 = __expf(S[1][r] * scale - SMAX);
            lsum[r] += p0 + p1;
            const int Mr = r + 8 * hs;
            ldsP[wave][Mr * 32 + m]      = (_Float16)p0;
            ldsP[wave][Mr * 32 + 16 + m] = (_Float16)p1;
        }
        asm volatile("s_wait_dscnt 0" ::: "memory");  // own P stores visible

        // P as A-fragment (16 q-rows x 32 keys); batch V^T fragment loads
        v16h aP;
        {
            const int ca = hs * 16;
            aP = *(const v16h*)(&ldsP[wave][m * 32 + ca]);
        }
        v16h bV[4];
        #pragma unroll
        for (int c = 0; c < 4; ++c)
            bV[c] = *(const v16h*)(&ldsVt[cur][(c * 16 + m) * 32 + hs * 16]);
        #pragma unroll
        for (int c = 0; c < 4; ++c)
            accO[c] = __builtin_amdgcn_wmma_f32_16x16x32_f16(
                false, aP, false, bV[c], (short)0, accO[c], false, false);

        __syncthreads();                              // done before buf reuse
    }

    // One final cross-lane reduction of the row sums (per half-wave row set)
    float linv[8];
    #pragma unroll
    for (int r = 0; r < 8; ++r) {
        float s = lsum[r];
        #pragma unroll
        for (int off = 8; off >= 1; off >>= 1) s += __shfl_xor(s, off, 32);
        linv[r] = 1.0f / s;
    }

    // Store in (B, L, H*Dh) layout (no transpose pass needed)
    #pragma unroll
    for (int c = 0; c < 4; ++c) {
        #pragma unroll
        for (int r = 0; r < 8; ++r) {
            const int Mr = r + 8 * hs;
            const float o = accO[c][r] * linv[r];
            O[(rowBase + q0 + Mr) * D_MODEL + h * HEAD_DIM + c * 16 + m] =
                (_Float16)o;
        }
    }
}

// ---------------------------------------------------------------------------
extern "C" void kernel_launch(void* const* d_in, const int* in_sizes, int n_in,
                              void* d_out, int out_size, void* d_ws,
                              size_t ws_size, hipStream_t stream) {
    (void)in_sizes; (void)n_in; (void)out_size; (void)ws_size;
    const float* x  = (const float*)d_in[0];
    const float* Wq = (const float*)d_in[1];
    const float* bq = (const float*)d_in[2];
    const float* Wk = (const float*)d_in[3];
    const float* bk = (const float*)d_in[4];
    const float* Wv = (const float*)d_in[5];
    const float* bv = (const float*)d_in[6];
    const float* Wo = (const float*)d_in[7];
    const float* bo = (const float*)d_in[8];
    const float* gq = (const float*)d_in[9];
    const float* gk = (const float*)d_in[10];
    float* out = (float*)d_out;

    char* ws = (char*)d_ws;
    const size_t MB = (size_t)1 << 20;
    _Float16* x16  = (_Float16*)(ws +  0 * MB);   // 8 MB   [4096,1024]
    _Float16* Wq16 = (_Float16*)(ws +  8 * MB);   // 2 MB
    _Float16* Wk16 = (_Float16*)(ws + 10 * MB);   // 2 MB
    _Float16* Wv16 = (_Float16*)(ws + 12 * MB);   // 2 MB
    _Float16* Wo16 = (_Float16*)(ws + 14 * MB);   // 2 MB
    _Float16* Q16  = (_Float16*)(ws + 16 * MB);   // 8 MB  [4096,1024]
    _Float16* K16  = (_Float16*)(ws + 24 * MB);   // 8 MB  [4096,1024]
    _Float16* Vt16 = (_Float16*)(ws + 32 * MB);   // 8 MB  [1024,4096] (V^T)
    _Float16* A16  = (_Float16*)(ws + 40 * MB);   // 8 MB  [4096,1024]

    const int nx = ROWS * D_MODEL;                // 4,194,304
    const int nw = D_MODEL * D_MODEL;             // 1,048,576
    cvt_f32_to_f16<<<(nx + 255) / 256, 256, 0, stream>>>(x,  x16,  nx);
    cvt_f32_to_f16<<<(nw + 255) / 256, 256, 0, stream>>>(Wq, Wq16, nw);
    cvt_f32_to_f16<<<(nw + 255) / 256, 256, 0, stream>>>(Wk, Wk16, nw);
    cvt_f32_to_f16<<<(nw + 255) / 256, 256, 0, stream>>>(Wv, Wv16, nw);
    cvt_f32_to_f16<<<(nw + 255) / 256, 256, 0, stream>>>(Wo, Wo16, nw);

    // 256 M-tiles x 16 N-supertiles = 4096 wave-tiles, 8 waves/block
    // Q/K: projection + sfn + fused per-head RMS norm (MODE 3)
    wmma_gemm_bias<3><<<512, 256, 0, stream>>>(x16, Wq16, bq, gq, Q16,  nullptr, ROWS);
    wmma_gemm_bias<3><<<512, 256, 0, stream>>>(x16, Wk16, bk, gk, K16,  nullptr, ROWS);
    // V: projection + sfn, stored transposed (MODE 1)
    wmma_gemm_bias<1><<<512, 256, 0, stream>>>(x16, Wv16, bv, nullptr, Vt16, nullptr, ROWS);

    // 16 q-supertiles x 32 (b,h) pairs; 8 waves share K/V^T chunks
    flash_attn_kernel<<<dim3(SEQ / 128, BATCH * NHEAD), 256, 0, stream>>>(
        Q16, K16, Vt16, A16);

    // Output projection, f32 (MODE 2)
    wmma_gemm_bias<2><<<512, 256, 0, stream>>>(A16, Wo16, bo, nullptr, nullptr, out, ROWS);
}
